// MultiHeadAttention_88553635709391
// MI455X (gfx1250) — compile-verified, matrix operand order checked by reading, not run on hardware
//
#include <hip/hip_runtime.h>

// ---------------------------------------------------------------------------
// MultiHeadAttention for MI455X (gfx1250, wave32, WMMA 16x16x32 f16 -> f32)
//
// Pipeline:
//   0. cvt:  f32 -> f16 staging of activations + weights (bandwidth trivial)
//   1. Q = (x Wq^T + bq) * d^-0.5  -> f16 [B][SQ][HIDDEN]
//   2. K = (x Wk^T + bk)           -> f16 [B][SK][HIDDEN]
//   3. Vt = (x Wv^T + bv)^T        -> f16 [B][HIDDEN][SK] (per-head transposed)
//   4. ctx = flash attention       -> f16 [B][SQ][HIDDEN]
//   5. out = ctx Wo^T + bo         -> f32 d_out
//
// GEMM: block tile 128x256 double-buffered through LDS with
// GLOBAL_LOAD_ASYNC_TO_LDS_B128 (ASYNCcnt), 64x64 tile per wave.
// Flash: 64-key inner blocks (16 wmma per softmax pass), DPP16 butterfly
// row reductions, P transposed through a bank-skewed LDS tile.
// ---------------------------------------------------------------------------

#define HIDDEN   1024
#define HEADS    16
#define HEAD_DIM 64
#define BATCH    4
#define SQ       2048
#define SK       2048

typedef __attribute__((ext_vector_type(16))) _Float16 v16h;
typedef __attribute__((ext_vector_type(8)))  _Float16 v8h;
typedef __attribute__((ext_vector_type(8)))  float    v8f;

// D = A(16x32 f16) * B(32x16 f16) + C(16x16 f32)
__device__ __forceinline__ v8f wmma32(v16h a, v16h b, v8f c) {
    return __builtin_amdgcn_wmma_f32_16x16x32_f16(
        false, a, false, b, (short)0, c, false, false);
}

// 16-bit A/B fragment packing (wave32): lane l holds row/col (l%16); element e
// has k = (e/8)*16 + (l/16)*8 + (e%8) -> two contiguous 16-byte runs per lane.
__device__ __forceinline__ v16h make_frag(const _Float16* __restrict__ p, int hi) {
    const v8h lo = *(const v8h*)(p + hi * 8);
    const v8h hs = *(const v8h*)(p + hi * 8 + 16);
    return __builtin_shufflevector(lo, hs, 0,1,2,3,4,5,6,7,8,9,10,11,12,13,14,15);
}

// ---- async global -> LDS copy (16B per lane, ASYNCcnt-tracked) -------------
__device__ __forceinline__ void async_copy16(void* lds_p, const void* gp) {
    const uint32_t loff = (uint32_t)(uintptr_t)lds_p;   // LDS segment offset
    asm volatile("global_load_async_to_lds_b128 %0, %1, off"
                 :: "v"(loff), "v"(gp) : "memory");
}
__device__ __forceinline__ void async_wait0() {
    asm volatile("s_wait_asynccnt 0" ::: "memory");
}

// ---- DPP16 row (16-lane) butterfly reductions ------------------------------
template <int CTRL>
__device__ __forceinline__ float dppf(float x) {
    int i = __builtin_bit_cast(int, x);
    i = __builtin_amdgcn_update_dpp(0, i, CTRL, 0xF, 0xF, true);
    return __builtin_bit_cast(float, i);
}
__device__ __forceinline__ float row_max16(float x) {
    x = fmaxf(x, dppf<0xB1>(x));    // quad_perm(1,0,3,2)  : xor 1
    x = fmaxf(x, dppf<0x4E>(x));    // quad_perm(2,3,0,1)  : xor 2
    x = fmaxf(x, dppf<0x141>(x));   // row_half_mirror
    x = fmaxf(x, dppf<0x140>(x));   // row_mirror
    return x;
}
__device__ __forceinline__ float row_sum16(float x) {
    x += dppf<0xB1>(x);
    x += dppf<0x4E>(x);
    x += dppf<0x141>(x);
    x += dppf<0x140>(x);
    return x;
}

// ---------------------------------------------------------------------------
// f32 -> f16 staging (8 elements / thread)
// ---------------------------------------------------------------------------
__global__ __launch_bounds__(256) void cvt_f16_kernel(
    const float* __restrict__ in, _Float16* __restrict__ out, int n8)
{
    const int i = blockIdx.x * blockDim.x + threadIdx.x;
    if (i >= n8) return;
    const float4 a = ((const float4*)in)[i * 2];
    const float4 b = ((const float4*)in)[i * 2 + 1];
    v8h o;
    o[0] = (_Float16)a.x; o[1] = (_Float16)a.y;
    o[2] = (_Float16)a.z; o[3] = (_Float16)a.w;
    o[4] = (_Float16)b.x; o[5] = (_Float16)b.y;
    o[6] = (_Float16)b.z; o[7] = (_Float16)b.w;
    ((v8h*)out)[i] = o;
}

// ---------------------------------------------------------------------------
// GEMM with bias, all-f16 operands: out = (A[MxK] * W[NxK]^T + bias) * scale
//
// 8 waves / 256-thread block; block tile 128(M) x 256(N); wave tile 64x64.
// Per K-step of 32 both operand panels are staged into LDS once via
// global_load_async_to_lds_b128 (24 KB unique data, double buffered),
// then waves read WMMA fragments back with ds_load_b128.
// LDS row stride = 40 halves (80 B): 16B-aligned for b128 and bank-skewed.
//
// MODE 0: out[m*N + n]                              (row-major, type OT)
// MODE 1: out[((m/SQdim)*N + n)*SQdim + m%SQdim]    (per-batch transposed)
// ---------------------------------------------------------------------------
#define LDS_STRIDE 40                       // halves per 32-half row (padded)
#define A_ROWS     128
#define W_ROWS     256

template <typename OT, int MODE>
__global__ __launch_bounds__(256) void gemm_f16_kernel(
    const _Float16* __restrict__ A, const _Float16* __restrict__ W,
    const float* __restrict__ bias, OT* __restrict__ out,
    int M, int N, int K, int SQdim, float scale)
{
    __shared__ __align__(16) _Float16 lds[2][(A_ROWS + W_ROWS) * LDS_STRIDE];

    const int tid  = threadIdx.x;
    const int lane = tid & 31;
    const int wave = tid >> 5;
    const int wm   = wave & 1;          // 2 waves along M
    const int wn   = wave >> 1;         // 4 waves along N
    const int ln   = lane & 15;
    const int hi   = lane >> 4;

    const int bm0 = blockIdx.x * 128;
    const int bn0 = blockIdx.y * 256;
    const int bm  = bm0 + wm * 64;
    const int bn  = bn0 + wn * 64;

    // Stage one 32-wide K-panel of A (128 rows) and W (256 rows) into LDS.
    auto stage = [&](int buf, int kb) {
        const _Float16* Ag = A + (size_t)bm0 * K + kb;
        const _Float16* Wg = W + (size_t)bn0 * K + kb;
        _Float16* Al = &lds[buf][0];
        _Float16* Wl = &lds[buf][A_ROWS * LDS_STRIDE];
#pragma unroll
        for (int c = 0; c < 2; ++c) {               // 128 rows * 4 segs / 256
            const int idx = c * 256 + tid;
            const int row = idx >> 2, seg = idx & 3;
            async_copy16(Al + row * LDS_STRIDE + seg * 8,
                         Ag + (size_t)row * K + seg * 8);
        }
#pragma unroll
        for (int c = 0; c < 4; ++c) {               // 256 rows * 4 segs / 256
            const int idx = c * 256 + tid;
            const int row = idx >> 2, seg = idx & 3;
            async_copy16(Wl + row * LDS_STRIDE + seg * 8,
                         Wg + (size_t)row * K + seg * 8);
        }
    };

    v8f acc[4][4] = {};

    stage(0, 0);
    int cur = 0;
    for (int kb = 0; kb < K; kb += 32, cur ^= 1) {
        async_wait0();          // my copies for this stage have landed
        __syncthreads();        // everyone's have
        if (kb + 32 < K) stage(cur ^ 1, kb + 32);

        const _Float16* Al = &lds[cur][0];
        const _Float16* Wl = &lds[cur][A_ROWS * LDS_STRIDE];

        v16h af[4], bf[4];
#pragma unroll
        for (int i = 0; i < 4; ++i)
            af[i] = make_frag(Al + (wm * 64 + i * 16 + ln) * LDS_STRIDE, hi);
#pragma unroll
        for (int j = 0; j < 4; ++j)
            bf[j] = make_frag(Wl + (wn * 64 + j * 16 + ln) * LDS_STRIDE, hi);
#pragma unroll
        for (int i = 0; i < 4; ++i)
#pragma unroll
            for (int j = 0; j < 4; ++j)
                acc[i][j] = wmma32(af[i], bf[j], acc[i][j]);

        __syncthreads();        // done reading buf before it is re-staged
    }

#pragma unroll
    for (int j = 0; j < 4; ++j) {
        const int   n  = bn + j * 16 + ln;
        const float bj = bias[n];
#pragma unroll
        for (int i = 0; i < 4; ++i) {
#pragma unroll
            for (int r = 0; r < 8; ++r) {
                const int   m = bm + i * 16 + r + 8 * hi;
                const float v = (acc[i][j][r] + bj) * scale;
                if (MODE == 0) {
                    out[(size_t)m * N + n] = (OT)v;
                } else {
                    const int bi = m / SQdim, s = m % SQdim;
                    out[((size_t)(bi * N + n)) * SQdim + s] = (OT)v;
                }
            }
        }
    }
}

// ---------------------------------------------------------------------------
// Flash attention: one wave per 16-query tile; online softmax over 64-key
// blocks (16 wmma per softmax pass).  Q pre-scaled by HEAD_DIM^-0.5.
//   Q,K : f16 [B][S][HIDDEN]    (head h at column h*64)
//   Vt  : f16 [B][HIDDEN][SK]   (per-head transposed so PV B-frags stream)
//   ctx : f16 [B][SQ][HIDDEN]
//
// P staging tile: 16 rows x 64 cols, row stride 72 halves (144 B): 16B-
// aligned for b128 reads, and the 36-dword row skew puts all 16 reading
// lanes in distinct bank groups.
// ---------------------------------------------------------------------------
#define P_STRIDE 72

__global__ __launch_bounds__(128) void flash_attn_kernel(
    const _Float16* __restrict__ Q, const _Float16* __restrict__ Kh,
    const _Float16* __restrict__ Vt, _Float16* __restrict__ ctx)
{
    __shared__ __align__(16) _Float16 plds[4][16 * P_STRIDE];  // per wave

    const int lane = threadIdx.x & 31;
    const int wave = threadIdx.x >> 5;
    const int ln   = lane & 15;
    const int hi   = lane >> 4;

    const int bh = blockIdx.y;
    const int b  = bh >> 4;
    const int h  = bh & 15;
    const int qbase = (blockIdx.x * 4 + wave) * 16;

    const size_t qrow = ((size_t)(b * SQ + qbase + ln)) * HIDDEN + h * HEAD_DIM;
    const v16h qf0 = make_frag(Q + qrow, hi);
    const v16h qf1 = make_frag(Q + qrow + 32, hi);

    float mrow[8], lrow[8];
    v8f   acc[4] = {};
#pragma unroll
    for (int r = 0; r < 8; ++r) { mrow[r] = -1e30f; lrow[r] = 0.0f; }

    _Float16* pw = &plds[wave][0];

    for (int kb = 0; kb < SK; kb += 64) {
        // ---- scores: S(16 x 64) as four 16x16 tiles, k-dim = 64 ------------
        v8f sc[4];
#pragma unroll
        for (int t = 0; t < 4; ++t) {
            const size_t krow =
                ((size_t)(b * SK + kb + t * 16 + ln)) * HIDDEN + h * HEAD_DIM;
            const v16h kf0 = make_frag(Kh + krow, hi);
            const v16h kf1 = make_frag(Kh + krow + 32, hi);
            v8f c = {};
            c = wmma32(qf0, kf0, c);
            c = wmma32(qf1, kf1, c);
            sc[t] = c;
        }

        // ---- online softmax (DPP row reductions over the 16-lane groups) --
#pragma unroll
        for (int r = 0; r < 8; ++r) {
            const float mb = row_max16(fmaxf(fmaxf(sc[0][r], sc[1][r]),
                                             fmaxf(sc[2][r], sc[3][r])));
            const float mn = fmaxf(mrow[r], mb);
            const float cf = __expf(mrow[r] - mn);   // accumulator rescale
            mrow[r] = mn;
            float ps = 0.0f;
#pragma unroll
            for (int t = 0; t < 4; ++t) {
                const float p = __expf(sc[t][r] - mn);
                sc[t][r] = p;
                ps += p;
            }
            lrow[r] = lrow[r] * cf + row_sum16(ps);
#pragma unroll
            for (int d = 0; d < 4; ++d) acc[d][r] *= cf;
        }

        // ---- transpose P(16x64) through LDS into WMMA-A fragments ----------
#pragma unroll
        for (int t = 0; t < 4; ++t)
#pragma unroll
            for (int r = 0; r < 8; ++r)
                pw[(r + 8 * hi) * P_STRIDE + t * 16 + ln] = (_Float16)sc[t][r];
        asm volatile("s_wait_dscnt 0" ::: "memory");
        const v16h pf0 = make_frag(pw + ln * P_STRIDE, hi);        // keys 0-31
        const v16h pf1 = make_frag(pw + ln * P_STRIDE + 32, hi);   // keys 32-63

        // ---- PV: ctx(16 x 64) += P(16x64) * V(64x64) -----------------------
#pragma unroll
        for (int d = 0; d < 4; ++d) {
            const size_t vrow =
                ((size_t)(b * HIDDEN + h * HEAD_DIM + d * 16 + ln)) * SK + kb;
            const v16h vb0 = make_frag(Vt + vrow, hi);
            const v16h vb1 = make_frag(Vt + vrow + 32, hi);
            acc[d] = wmma32(pf0, vb0, acc[d]);
            acc[d] = wmma32(pf1, vb1, acc[d]);
        }
    }

    // ---- normalize and store ctx (f16) ------------------------------------
#pragma unroll
    for (int d = 0; d < 4; ++d) {
#pragma unroll
        for (int r = 0; r < 8; ++r) {
            const int   m = qbase + r + 8 * hi;
            const float v = acc[d][r] / lrow[r];
            ctx[((size_t)(b * SQ + m)) * HIDDEN + h * HEAD_DIM + d * 16 + ln] =
                (_Float16)v;
        }
    }
}

// ---------------------------------------------------------------------------
extern "C" void kernel_launch(void* const* d_in, const int* in_sizes, int n_in,
                              void* d_out, int out_size, void* d_ws, size_t ws_size,
                              hipStream_t stream) {
    const float* query = (const float*)d_in[0];
    const float* key   = (const float*)d_in[1];
    const float* value = (const float*)d_in[2];
    const float* Wq    = (const float*)d_in[3];
    const float* bq    = (const float*)d_in[4];
    const float* Wk    = (const float*)d_in[5];
    const float* bk    = (const float*)d_in[6];
    const float* Wv    = (const float*)d_in[7];
    const float* bv    = (const float*)d_in[8];
    const float* Wo    = (const float*)d_in[9];
    const float* bo    = (const float*)d_in[10];
    float*       out   = (float*)d_out;

    const size_t tok  = (size_t)BATCH * SQ;            // 8192 tokens
    const size_t act  = tok * HIDDEN;                  // 8M elements
    const size_t wsz  = (size_t)HIDDEN * HIDDEN;       // 1M elements

    // ws layout (f16 elements); ctx aliases xq (dead after Q projection).
    _Float16* base = (_Float16*)d_ws;
    _Float16* xq  = base;                 // act
    _Float16* xk  = xq  + act;            // act
    _Float16* xv  = xk  + act;            // act
    _Float16* wqh = xv  + act;            // wsz
    _Float16* wkh = wqh + wsz;
    _Float16* wvh = wkh + wsz;
    _Float16* woh = wvh + wsz;
    _Float16* Qp  = woh + wsz;            // act
    _Float16* Kp  = Qp  + act;            // act
    _Float16* Vtp = Kp  + act;            // act
    _Float16* ctx = xq;                   // alias  (peak ~104 MB)

    // ---- stage f32 -> f16 --------------------------------------------------
    {
        const int a8 = (int)(act / 8), w8 = (int)(wsz / 8);
        cvt_f16_kernel<<<a8 / 256, 256, 0, stream>>>(query, xq, a8);
        cvt_f16_kernel<<<a8 / 256, 256, 0, stream>>>(key,   xk, a8);
        cvt_f16_kernel<<<a8 / 256, 256, 0, stream>>>(value, xv, a8);
        cvt_f16_kernel<<<w8 / 256, 256, 0, stream>>>(Wq, wqh, w8);
        cvt_f16_kernel<<<w8 / 256, 256, 0, stream>>>(Wk, wkh, w8);
        cvt_f16_kernel<<<w8 / 256, 256, 0, stream>>>(Wv, wvh, w8);
        cvt_f16_kernel<<<w8 / 256, 256, 0, stream>>>(Wo, woh, w8);
    }

    const dim3 gblk((unsigned)(tok / 128), HIDDEN / 256);   // 64 x 4 blocks
    const float qscale = 0.125f;                            // HEAD_DIM^-0.5

    gemm_f16_kernel<_Float16, 0><<<gblk, 256, 0, stream>>>(
        xq, wqh, bq, Qp, (int)tok, HIDDEN, HIDDEN, SQ, qscale);
    gemm_f16_kernel<_Float16, 0><<<gblk, 256, 0, stream>>>(
        xk, wkh, bk, Kp, (int)tok, HIDDEN, HIDDEN, SK, 1.0f);
    gemm_f16_kernel<_Float16, 1><<<gblk, 256, 0, stream>>>(
        xv, wvh, bv, Vtp, (int)tok, HIDDEN, HIDDEN, SK, 1.0f);

    flash_attn_kernel<<<dim3(SQ / 16 / 4, BATCH * HEADS), 128, 0, stream>>>(
        Qp, Kp, Vtp, ctx);

    gemm_f16_kernel<float, 0><<<gblk, 256, 0, stream>>>(
        ctx, woh, bo, out, (int)tok, HIDDEN, HIDDEN, SQ, 1.0f);
}
